// GatedMultiheadAttention_18966575579157
// MI455X (gfx1250) — matrix operand order repaired and checked by reading, not compiled, not run on hardware
//
#include <hip/hip_runtime.h>
#include <math.h>

// GatedMultiheadAttention for MI455X (gfx1250), wave32 + bf16 WMMA.
// B=4, L=1024, E=1024, H=16, D=64, HID=1024, R=4.

#define BB  4
#define LL  1024
#define EE  1024
#define HH  16
#define DDIM 64
#define HID 1024
#define TOK (BB*LL)          // 4096 tokens

typedef unsigned short u16;
typedef __attribute__((ext_vector_type(16))) __bf16 v16bf;
typedef __attribute__((ext_vector_type(8)))  float  v8f;
typedef __attribute__((ext_vector_type(4)))  float  f32x4;

struct BPair { uint4 lo; uint4 hi; };   // 32 bytes == one v16bf operand

static __device__ __forceinline__ u16 f2bf(float f) {
  unsigned int u = __float_as_uint(f);
  u += 0x7FFFu + ((u >> 16) & 1u);      // round-to-nearest-even
  return (u16)(u >> 16);
}

// A operand (16x32 bf16, M=lane&15): lanes 0-15 hold K = {0..7, 16..23},
// lanes 16-31 hold K = {8..15, 24..31}.  row points at (row start + kk).
static __device__ __forceinline__ v16bf load_a_tile(const u16* row, int laneHi) {
  BPair p;
  p.lo = *(const uint4*)(row + laneHi * 8);
  p.hi = *(const uint4*)(row + 16 + laneHi * 8);
  return __builtin_bit_cast(v16bf, p);
}

// B operand (32x16 bf16, N=lane&15): lanes 0-15 hold K = {0..7, 16..23},
// lanes 16-31 hold K = {8..15, 24..31} (same order as the A operand).
static __device__ __forceinline__ v16bf load_b_tile(const u16* row, int laneHi) {
  BPair p;
  p.lo = *(const uint4*)(row + laneHi * 8);
  p.hi = *(const uint4*)(row + 16 + laneHi * 8);
  return __builtin_bit_cast(v16bf, p);
}

static __device__ __forceinline__ v8f wmma_bf16(v16bf a, v16bf b, v8f c) {
  return __builtin_amdgcn_wmma_f32_16x16x32_bf16(false, a, false, b, (short)0, c,
                                                 false, false);
}

// Ping-pong pipelined 16x64 GEMM body, K=1024, all row strides 1024 elements.
// B rows for the 4 N-tiles are b0 + nt*16*1024.  Two 32-wide K stages whose
// registers are refilled IN PLACE after use (no buffer-rotate v_movs).
template <bool PF_A>
static __device__ __forceinline__ void gemm_16x64_k1024(
    const u16* __restrict__ arow, const u16* __restrict__ b0,
    int laneHi, v8f acc[4]) {
  v16bf aA = load_a_tile(arow, laneHi);
  v16bf bA[4];
#pragma unroll
  for (int nt = 0; nt < 4; ++nt) bA[nt] = load_b_tile(b0 + nt * 16384, laneHi);
  v16bf aB = load_a_tile(arow + 32, laneHi);
  v16bf bB[4];
#pragma unroll
  for (int nt = 0; nt < 4; ++nt) bB[nt] = load_b_tile(b0 + nt * 16384 + 32, laneHi);

  for (int kk = 64; kk < 1024; kk += 64) {
    if (PF_A) __builtin_prefetch(arow + kk + 448, 0, 0);  // ~1KB ahead
#pragma unroll
    for (int nt = 0; nt < 4; ++nt) acc[nt] = wmma_bf16(aA, bA[nt], acc[nt]);
    aA = load_a_tile(arow + kk, laneHi);
#pragma unroll
    for (int nt = 0; nt < 4; ++nt)
      bA[nt] = load_b_tile(b0 + nt * 16384 + kk, laneHi);
#pragma unroll
    for (int nt = 0; nt < 4; ++nt) acc[nt] = wmma_bf16(aB, bB[nt], acc[nt]);
    aB = load_a_tile(arow + kk + 32, laneHi);
#pragma unroll
    for (int nt = 0; nt < 4; ++nt)
      bB[nt] = load_b_tile(b0 + nt * 16384 + kk + 32, laneHi);
  }
#pragma unroll
  for (int nt = 0; nt < 4; ++nt) acc[nt] = wmma_bf16(aA, bA[nt], acc[nt]);
#pragma unroll
  for (int nt = 0; nt < 4; ++nt) acc[nt] = wmma_bf16(aB, bB[nt], acc[nt]);
}

// ---------------------------------------------------------------- prep: f32->bf16
__global__ void __launch_bounds__(256) k_prep(const float* __restrict__ query,
                                              const float* __restrict__ proj_w,
                                              const float* __restrict__ out_w,
                                              u16* __restrict__ qx,
                                              u16* __restrict__ wqkvg,
                                              u16* __restrict__ wout) {
  const int NQ = TOK * EE;          // 4M
  const int NW = 4 * HID * EE;      // 4M
  const int NO = EE * HID;          // 1M
  int stride = gridDim.x * blockDim.x;
  for (int i = blockIdx.x * blockDim.x + threadIdx.x; i < NQ + NW + NO; i += stride) {
    if (i < NQ)            qx[i]             = f2bf(query[i]);
    else if (i < NQ + NW)  wqkvg[i - NQ]     = f2bf(proj_w[i - NQ]);
    else                   wout[i - NQ - NW] = f2bf(out_w[i - NQ - NW]);
  }
}

// ---------------------------------------------------- GEMM0: qkvg = X @ Wqkvg^T + b
// One wave: 16 tokens x 64 output cols.  Routes q,k (bf16), tanh(v) transposed
// to [b,h,d,j] (bf16), sigmoid(g) (f32).
__global__ void __launch_bounds__(256) k_gemm_qkvg(const u16* __restrict__ qx,
                                                   const u16* __restrict__ wqkvg,
                                                   const float* __restrict__ proj_b,
                                                   u16* __restrict__ qb,
                                                   u16* __restrict__ kb,
                                                   u16* __restrict__ tvT,
                                                   float* __restrict__ sg) {
  int wave   = blockIdx.x * (blockDim.x >> 5) + (threadIdx.x >> 5); // 0..16383
  int mt     = wave >> 6;          // 256 token tiles
  int ns     = wave & 63;          // 64 col strips (64 cols each)
  int lane   = threadIdx.x & 31;
  int laneHi = lane >> 4, lanelo = lane & 15;
  int t0 = mt * 16, n0 = ns * 64;

  v8f acc[4] = {};
  gemm_16x64_k1024<false>(qx + (t0 + lanelo) * EE,
                          wqkvg + (n0 + lanelo) * EE, laneHi, acc);
#pragma unroll
  for (int nt = 0; nt < 4; ++nt) {
    int c    = n0 + nt * 16 + lanelo;     // output column (N = lane&15)
    float bv = proj_b[c];
    int sec  = c >> 10, cc = c & 1023;
#pragma unroll
    for (int p = 0; p < 8; ++p) {
      int t   = t0 + p + laneHi * 8;      // token (M = p + 8*laneHi)
      float v = acc[nt][p] + bv;
      if (sec == 0)      qb[t * HID + cc] = f2bf(v);
      else if (sec == 1) kb[t * HID + cc] = f2bf(v);
      else if (sec == 2) {
        int b_ = t >> 10, i = t & 1023, h = cc >> 6, d = cc & 63;
        tvT[(((b_ * HH + h) * DDIM + d) << 10) + i] = f2bf(tanhf(v));
      } else {
        sg[t * HID + cc] = 1.0f / (1.0f + __expf(-v));
      }
    }
  }
}

// ------------------- attention: scores for ALL 16 heads of one 16x16 (i,j) tile,
// rels bias, scale, log-softmax over HEADS (per-lane!), +mask, exp, key padding.
// Writes w f32 [b,i,j,h] (non-temporal) to d_out and w bf16 [b,h,i,j] to ws.
__global__ void __launch_bounds__(256) k_attn(const u16* __restrict__ qb,
                                              const u16* __restrict__ kb,
                                              const float* __restrict__ rels,
                                              const float* __restrict__ attn_mask,
                                              const unsigned char* __restrict__ kpm,
                                              const float* __restrict__ rels_bias,
                                              float* __restrict__ w_out,
                                              u16* __restrict__ wb) {
  int wave   = blockIdx.x * (blockDim.x >> 5) + (threadIdx.x >> 5); // 0..16383
  int b      = wave >> 12;
  int it     = (wave >> 6) & 63;
  int jt     = wave & 63;
  int lane   = threadIdx.x & 31;
  int laneHi = lane >> 4, lanelo = lane & 15;
  int i0 = it * 16, j0 = jt * 16;
  int j  = j0 + lanelo;

  // Hoist the HBM-streamed epilogue inputs so their latency hides under the
  // 32-WMMA chain below (rels 67MB + mask 17MB + kpm stream from HBM).
  f32x4 r4[8]; float am[8]; unsigned char pd[8];
#pragma unroll
  for (int p = 0; p < 8; ++p) {
    int i  = i0 + p + laneHi * 8;
    int ij = (b * LL + i) * LL + j;
    r4[p] = *(const f32x4*)(rels + (size_t)ij * 4);
    am[p] = attn_mask[ij];
    pd[p] = kpm[ij];
  }

  // 16 heads of the same 16x16 (i,j) tile -> per-(i,j) head values land in the
  // same lane/slot across the 16 accumulators (softmax over h is per-lane VALU).
  // Ping-pong over head pairs: operand sets refilled in place, no rotate movs.
  const u16* aR = qb + ((b * LL + i0 + lanelo) * HH) * DDIM;  // +h*64
  const u16* bR = kb + ((b * LL + j0 + lanelo) * HH) * DDIM;  // +h*64
  v8f acc[16] = {};
  v16bf a0A = load_a_tile(aR, laneHi),          a1A = load_a_tile(aR + 32, laneHi);
  v16bf b0A = load_b_tile(bR, laneHi),          b1A = load_b_tile(bR + 32, laneHi);
  v16bf a0B = load_a_tile(aR + DDIM, laneHi),   a1B = load_a_tile(aR + DDIM + 32, laneHi);
  v16bf b0B = load_b_tile(bR + DDIM, laneHi),   b1B = load_b_tile(bR + DDIM + 32, laneHi);
#pragma unroll
  for (int h = 0; h < HH; h += 2) {
    acc[h] = wmma_bf16(a0A, b0A, acc[h]);
    acc[h] = wmma_bf16(a1A, b1A, acc[h]);
    if (h + 2 < HH) {           // compile-time resolved (fully unrolled)
      const u16* an = aR + (h + 2) * DDIM;
      const u16* bn = bR + (h + 2) * DDIM;
      a0A = load_a_tile(an, laneHi); a1A = load_a_tile(an + 32, laneHi);
      b0A = load_b_tile(bn, laneHi); b1A = load_b_tile(bn + 32, laneHi);
    }
    acc[h + 1] = wmma_bf16(a0B, b0B, acc[h + 1]);
    acc[h + 1] = wmma_bf16(a1B, b1B, acc[h + 1]);
    if (h + 3 < HH) {
      const u16* an = aR + (h + 3) * DDIM;
      const u16* bn = bR + (h + 3) * DDIM;
      a0B = load_a_tile(an, laneHi); a1B = load_a_tile(an + 32, laneHi);
      b0B = load_b_tile(bn, laneHi); b1B = load_b_tile(bn + 32, laneHi);
    }
  }

  // rels_bias is wave-uniform -> scalar loads
  float rb[4][16];
#pragma unroll
  for (int r = 0; r < 4; ++r)
#pragma unroll
    for (int h = 0; h < HH; ++h) rb[r][h] = rels_bias[r * HH + h];

  const float scale = 0.125f;   // D^-0.5, D=64
  int wbbase = (b * HH) * (LL * LL);
#pragma unroll
  for (int p = 0; p < 8; ++p) {
    int i  = i0 + p + laneHi * 8;
    int ij = (b * LL + i) * LL + j;
    float pad = pd[p] ? 1.0f : 0.0f;

    float sc[16];
    float mx = -3.402823466e38f;
#pragma unroll
    for (int h = 0; h < HH; ++h) {
      float s = acc[h][p] + r4[p].x * rb[0][h] + r4[p].y * rb[1][h]
                          + r4[p].z * rb[2][h] + r4[p].w * rb[3][h];
      s *= scale;
      sc[h] = s;
      mx = fmaxf(mx, s);
    }
    float se = 0.0f;
#pragma unroll
    for (int h = 0; h < HH; ++h) se += __expf(sc[h] - mx);
    float ls = mx + __logf(se);

    float wv[16];
#pragma unroll
    for (int h = 0; h < HH; ++h) wv[h] = __expf(sc[h] - ls + am[p]) * pad;

    // w f32, h contiguous, 64B aligned; written once, never re-read on device
    // -> non-temporal so it doesn't evict the L2-resident q/k/tvT/wb set.
    float* wp = w_out + (size_t)ij * HH;
    __builtin_nontemporal_store(f32x4{wv[0],  wv[1],  wv[2],  wv[3]},  (f32x4*)(wp + 0));
    __builtin_nontemporal_store(f32x4{wv[4],  wv[5],  wv[6],  wv[7]},  (f32x4*)(wp + 4));
    __builtin_nontemporal_store(f32x4{wv[8],  wv[9],  wv[10], wv[11]}, (f32x4*)(wp + 8));
    __builtin_nontemporal_store(f32x4{wv[12], wv[13], wv[14], wv[15]}, (f32x4*)(wp + 12));

    int wo = wbbase + i * LL + j;
#pragma unroll
    for (int h = 0; h < HH; ++h) wb[wo + h * (LL * LL)] = f2bf(wv[h]);
  }
}

// ----------------- GEMM2: attn_out[b,h] = W[b,h] (LxL) @ tanhV[b,h] (LxD), fused
// sigmoid gating, pack gated bf16 [t, h*64+d].  A (wb, 128MB) streams -> prefetch.
__global__ void __launch_bounds__(256) k_gemm_av(const u16* __restrict__ wb,
                                                 const u16* __restrict__ tvT,
                                                 const float* __restrict__ sg,
                                                 u16* __restrict__ gated) {
  int wave   = blockIdx.x * (blockDim.x >> 5) + (threadIdx.x >> 5); // 0..4095
  int b      = wave >> 10;
  int h      = (wave >> 6) & 15;
  int it     = wave & 63;
  int lane   = threadIdx.x & 31;
  int laneHi = lane >> 4, lanelo = lane & 15;
  int i0 = it * 16;

  v8f acc[4] = {};
  gemm_16x64_k1024<true>(wb  + ((b * HH + h) * LL + i0 + lanelo) * LL,
                         tvT + ((b * HH + h) * DDIM + lanelo) * LL,
                         laneHi, acc);
#pragma unroll
  for (int nt = 0; nt < 4; ++nt) {
    int d = nt * 16 + lanelo;
#pragma unroll
    for (int p = 0; p < 8; ++p) {
      int i   = i0 + p + laneHi * 8;
      int t   = b * LL + i;
      int col = h * DDIM + d;
      float g = acc[nt][p] * sg[t * HID + col];
      gated[t * HID + col] = f2bf(g);
    }
  }
}

// ----------------- GEMM3: out = gated @ out_w^T + out_b  ->  d_out f32 (NT store)
__global__ void __launch_bounds__(256) k_gemm_out(const u16* __restrict__ gated,
                                                  const u16* __restrict__ wout,
                                                  const float* __restrict__ out_b,
                                                  float* __restrict__ out) {
  int wave   = blockIdx.x * (blockDim.x >> 5) + (threadIdx.x >> 5); // 0..4095
  int mt     = wave >> 4;          // 256 token tiles
  int ns     = wave & 15;          // 16 strips of 64 cols
  int lane   = threadIdx.x & 31;
  int laneHi = lane >> 4, lanelo = lane & 15;
  int t0 = mt * 16, n0 = ns * 64;

  v8f acc[4] = {};
  gemm_16x64_k1024<false>(gated + (t0 + lanelo) * HID,
                          wout + (n0 + lanelo) * HID, laneHi, acc);
#pragma unroll
  for (int nt = 0; nt < 4; ++nt) {
    int c = n0 + nt * 16 + lanelo;
    float bv = out_b[c];
#pragma unroll
    for (int p = 0; p < 8; ++p) {
      int t = t0 + p + laneHi * 8;
      __builtin_nontemporal_store(acc[nt][p] + bv, out + t * EE + c);
    }
  }
}

extern "C" void kernel_launch(void* const* d_in, const int* in_sizes, int n_in,
                              void* d_out, int out_size, void* d_ws, size_t ws_size,
                              hipStream_t stream) {
  const float*         query     = (const float*)d_in[0];
  const float*         rels      = (const float*)d_in[1];
  const float*         attn_mask = (const float*)d_in[2];
  const unsigned char* kpm       = (const unsigned char*)d_in[3];  // jnp bool -> 1B
  const float*         proj_w    = (const float*)d_in[4];
  const float*         proj_b    = (const float*)d_in[5];
  const float*         out_w     = (const float*)d_in[6];
  const float*         out_b     = (const float*)d_in[7];
  const float*         rels_bias = (const float*)d_in[8];

  float* out   = (float*)d_out;
  float* w_out = out + (size_t)TOK * EE;   // second tuple element [B,L,L,H]

  // workspace carve-up (~194 MiB)
  char* ws = (char*)d_ws;
  u16*   qx    = (u16*)ws;  ws += (size_t)TOK * EE * 2;        // 8 MiB
  u16*   wqkvg = (u16*)ws;  ws += (size_t)4 * HID * EE * 2;    // 8 MiB
  u16*   wout_ = (u16*)ws;  ws += (size_t)EE * HID * 2;        // 2 MiB
  u16*   qb    = (u16*)ws;  ws += (size_t)TOK * HID * 2;       // 8 MiB
  u16*   kb    = (u16*)ws;  ws += (size_t)TOK * HID * 2;       // 8 MiB
  u16*   tvT   = (u16*)ws;  ws += (size_t)TOK * HID * 2;       // 8 MiB
  float* sg    = (float*)ws; ws += (size_t)TOK * HID * 4;      // 16 MiB
  u16*   gated = (u16*)ws;  ws += (size_t)TOK * HID * 2;       // 8 MiB
  u16*   wb    = (u16*)ws;  /* BB*HH*LL*LL*2 = 128 MiB */

  k_prep<<<4096, 256, 0, stream>>>(query, proj_w, out_w, qx, wqkvg, wout_);
  k_gemm_qkvg<<<2048, 256, 0, stream>>>(qx, wqkvg, proj_b, qb, kb, tvT, sg);
  k_attn<<<2048, 256, 0, stream>>>(qb, kb, rels, attn_mask, kpm, rels_bias, w_out, wb);
  k_gemm_av<<<512, 256, 0, stream>>>(wb, tvT, sg, gated);
  k_gemm_out<<<512, 256, 0, stream>>>(gated, wout_, out_b, out);
}
